// RGCNLinkPredictor_48086453846017
// MI455X (gfx1250) — matrix operand order repaired and checked by reading, not compiled, not run on hardware
//
#include <hip/hip_runtime.h>

typedef __attribute__((ext_vector_type(16))) __bf16 v16bf;
typedef __attribute__((ext_vector_type(8)))  __bf16 v8bf;
typedef __attribute__((ext_vector_type(8)))  float  v8f;

#define N_ENT     100000
#define DIM       128
#define N_REL     8
#define E_PER_REL 200000
#define BATCH     8192
#define K_NEG     256
#define LN_EPS    1e-5f
#define ETILES    (E_PER_REL / 16)   // 12500
#define TILES_PER_BLK 8

__device__ __forceinline__ unsigned short f2bf(float f) {
  unsigned int u = __builtin_bit_cast(unsigned int, f);
  u += 0x7FFFu + ((u >> 16) & 1u);   // round-to-nearest-even
  return (unsigned short)(u >> 16);
}

// ---------------- bf16 conversion kernels ----------------

__global__ void cvt_x_bf16(const float* __restrict__ x, unsigned short* __restrict__ xb, int n) {
  int i = blockIdx.x * blockDim.x + threadIdx.x;
  if (i < n) xb[i] = f2bf(x[i]);
}

// wtb[(l*9+s)*16384 + n*128 + k] = bf16(W_{l,s}[k][n])   (transposed -> contiguous B-fragments)
__global__ void cvt_w_bf16(const float* __restrict__ selfW, const float* __restrict__ relW,
                           unsigned short* __restrict__ wtb) {
  int i = blockIdx.x * blockDim.x + threadIdx.x;       // 0 .. 2*9*16384-1
  int l   = i / (9 * 16384);
  int rem = i % (9 * 16384);
  int s   = rem / 16384;
  int p   = rem % 16384;
  int n = p >> 7, k = p & 127;
  float w = (s == 0) ? selfW[l * 16384 + k * 128 + n]
                     : relW[((l * 8) + (s - 1)) * 16384 + k * 128 + n];
  wtb[i] = f2bf(w);
}

// ---------------- WMMA GEMM kernels ----------------

// out[N,128] = x @ W   (W transposed bf16 128x128). Block: 8 waves = 8 N-tiles, 64 rows
// staged once in LDS, 4 M-tiles per block.
__global__ void self_gemm(const unsigned short* __restrict__ xb,
                          const unsigned short* __restrict__ wt,
                          float* __restrict__ out, int nrows) {
  __shared__ __align__(16) unsigned short lsa[64 * DIM];   // 16 KB
  const int lane = threadIdx.x & 31;
  const int wave = threadIdx.x >> 5;   // N-tile 0..7
  const int half = lane >> 4;
  const int lm   = lane & 15;
  const int m0   = blockIdx.x * 64;

  // B fragments (reused for all 4 M-tiles)
  v16bf bfrag[4];
  const unsigned short* wrow = wt + (size_t)(wave * 16 + lm) * DIM + half * 16;
#pragma unroll
  for (int ks = 0; ks < 4; ++ks)
    bfrag[ks] = *reinterpret_cast<const v16bf*>(wrow + ks * 32);

  // Cooperative stage of 64 A-rows into LDS (each thread copies 4x16B chunks)
#pragma unroll
  for (int it = 0; it < 4; ++it) {
    int task = threadIdx.x + it * 256;          // 0..1023 = 64 rows x 16 chunks
    int lrow = task >> 4, lchunk = task & 15;
    if (m0 + lrow < nrows) {
      const uint4 v = *reinterpret_cast<const uint4*>(xb + (size_t)(m0 + lrow) * DIM + lchunk * 8);
      *reinterpret_cast<uint4*>(lsa + lrow * DIM + lchunk * 8) = v;
    }
  }
  __syncthreads();

#pragma unroll
  for (int mt = 0; mt < 4; ++mt) {
    int m16 = m0 + mt * 16;
    if (m16 >= nrows) break;                    // uniform across block
    const unsigned short* arow = lsa + (mt * 16 + lm) * DIM;
    v8f acc = {};
#pragma unroll
    for (int ks = 0; ks < 4; ++ks) {
      union { v16bf v; v8bf h[2]; } a;
      a.h[0] = *reinterpret_cast<const v8bf*>(arow + ks * 32 + half * 16);
      a.h[1] = *reinterpret_cast<const v8bf*>(arow + ks * 32 + half * 16 + 8);
      acc = __builtin_amdgcn_wmma_f32_16x16x32_bf16(false, a.v, false, bfrag[ks],
                                                    (short)0, acc, false, false);
    }
    float* orow = out + (size_t)m16 * DIM + wave * 16 + lm;
#pragma unroll
    for (int v = 0; v < 8; ++v)
      orow[(size_t)(v + half * 8) * DIM] = acc[v];
  }
}

// msg = x[src] @ rel_W ; out[dst] += msg.
// Block: one relation (grid.y), 8 edge-tiles of 16 edges; A-tile staged in LDS per tile,
// B fragments hoisted and reused across all tiles.
__global__ void msg_gemm(const unsigned short* __restrict__ xb,
                         const unsigned short* __restrict__ wt_rel,  // layer rel base
                         const int* __restrict__ src, const int* __restrict__ dst,
                         float* __restrict__ out) {
  __shared__ __align__(16) unsigned short lsa[16 * DIM];   // 4 KB
  const int lane = threadIdx.x & 31;
  const int wave = threadIdx.x >> 5;
  const int half = lane >> 4;
  const int lm   = lane & 15;
  const int r    = blockIdx.y;
  const size_t rbase = (size_t)r * E_PER_REL;
  const unsigned short* wt = wt_rel + (size_t)r * 16384;

  // B fragments once per block
  v16bf bfrag[4];
  const unsigned short* wrow = wt + (size_t)(wave * 16 + lm) * DIM + half * 16;
#pragma unroll
  for (int ks = 0; ks < 4; ++ks)
    bfrag[ks] = *reinterpret_cast<const v16bf*>(wrow + ks * 32);

  for (int tt = 0; tt < TILES_PER_BLK; ++tt) {
    const int et = blockIdx.x * TILES_PER_BLK + tt;
    if (et >= ETILES) break;                    // uniform across block
    const int e0 = et * 16;

    __syncthreads();                            // previous tile's LDS reads done
    {
      const int lrow = threadIdx.x >> 4;        // 0..15
      const int lchunk = threadIdx.x & 15;      // 16B chunk
      const int srow = src[rbase + e0 + lrow];
      const uint4 v = *reinterpret_cast<const uint4*>(xb + (size_t)srow * DIM + lchunk * 8);
      *reinterpret_cast<uint4*>(lsa + lrow * DIM + lchunk * 8) = v;
    }
    __syncthreads();

    // destination rows for the scatter; prefetch them toward cache before compute
    int drow[8];
#pragma unroll
    for (int v = 0; v < 8; ++v) {
      drow[v] = dst[rbase + e0 + v + half * 8];
      __builtin_prefetch(out + (size_t)drow[v] * DIM + wave * 16, 1, 1);
    }

    const unsigned short* arow = lsa + lm * DIM;
    v8f acc = {};
#pragma unroll
    for (int ks = 0; ks < 4; ++ks) {
      union { v16bf v; v8bf h[2]; } a;
      a.h[0] = *reinterpret_cast<const v8bf*>(arow + ks * 32 + half * 16);
      a.h[1] = *reinterpret_cast<const v8bf*>(arow + ks * 32 + half * 16 + 8);
      acc = __builtin_amdgcn_wmma_f32_16x16x32_bf16(false, a.v, false, bfrag[ks],
                                                    (short)0, acc, false, false);
    }
#pragma unroll
    for (int v = 0; v < 8; ++v)
      atomicAdd(out + (size_t)drow[v] * DIM + wave * 16 + lm, acc[v]);
  }
}

// ---------------- LayerNorm + (ReLU) + residual, emits f32 next-x (in place over `io`) and bf16 copy

__global__ void ln_residual(const float* __restrict__ xcur, float* __restrict__ io,
                            unsigned short* __restrict__ xb,
                            const float* __restrict__ g, const float* __restrict__ b,
                            int nrows, int do_relu) {
  const int lane = threadIdx.x & 31;
  const int wave = threadIdx.x >> 5;
  const int row  = blockIdx.x * 8 + wave;
  if (row >= nrows) return;
  const float4 o  = reinterpret_cast<const float4*>(io   + (size_t)row * DIM)[lane];
  const float4 xc = reinterpret_cast<const float4*>(xcur + (size_t)row * DIM)[lane];
  float s = o.x + o.y + o.z + o.w;
#pragma unroll
  for (int m = 16; m >= 1; m >>= 1) s += __shfl_xor(s, m, 32);
  const float mu = s * (1.0f / DIM);
  const float dx = o.x - mu, dy = o.y - mu, dz = o.z - mu, dw = o.w - mu;
  float var = dx * dx + dy * dy + dz * dz + dw * dw;
#pragma unroll
  for (int m = 16; m >= 1; m >>= 1) var += __shfl_xor(var, m, 32);
  const float rs = rsqrtf(var * (1.0f / DIM) + LN_EPS);
  const float4 gg = reinterpret_cast<const float4*>(g)[lane];
  const float4 bb = reinterpret_cast<const float4*>(b)[lane];
  float h0 = dx * rs * gg.x + bb.x;
  float h1 = dy * rs * gg.y + bb.y;
  float h2 = dz * rs * gg.z + bb.z;
  float h3 = dw * rs * gg.w + bb.w;
  if (do_relu) {
    h0 = fmaxf(h0, 0.f); h1 = fmaxf(h1, 0.f); h2 = fmaxf(h2, 0.f); h3 = fmaxf(h3, 0.f);
  }
  const float4 xn = { xc.x + h0, xc.y + h1, xc.z + h2, xc.w + h3 };
  reinterpret_cast<float4*>(io + (size_t)row * DIM)[lane] = xn;
  union { unsigned short s4[4]; uint2 u; } pk;
  pk.s4[0] = f2bf(xn.x); pk.s4[1] = f2bf(xn.y); pk.s4[2] = f2bf(xn.z); pk.s4[3] = f2bf(xn.w);
  reinterpret_cast<uint2*>(xb + (size_t)row * DIM)[lane] = pk.u;
}

// ---------------- Decoder: pos + 2x256 negative scores per batch element ----------------

__global__ void decode_scores(const float* __restrict__ x, const float* __restrict__ rel_dec,
                              const int* __restrict__ h, const int* __restrict__ r,
                              const int* __restrict__ t,
                              const int* __restrict__ tneg, const int* __restrict__ hneg,
                              float* __restrict__ out) {
  const int b = blockIdx.x;
  const int tid = threadIdx.x;
  __shared__ __align__(16) float hr[DIM];
  __shared__ __align__(16) float rt[DIM];
  __shared__ float red[4];
  const int hb = h[b], rb = r[b], tb = t[b];
  if (tid < DIM) {
    const float xh = x[(size_t)hb * DIM + tid];
    const float rd = rel_dec[(size_t)rb * DIM + tid];
    const float xt = x[(size_t)tb * DIM + tid];
    const float a = xh * rd;
    hr[tid] = a;
    rt[tid] = rd * xt;
    float p = a * xt;
#pragma unroll
    for (int m = 16; m >= 1; m >>= 1) p += __shfl_xor(p, m, 32);
    if ((tid & 31) == 0) red[tid >> 5] = p;
  }
  __syncthreads();
  if (tid == 0) out[(size_t)b * 513] = red[0] + red[1] + red[2] + red[3];

  const float4* hr4 = reinterpret_cast<const float4*>(hr);
  const float4* rt4 = reinterpret_cast<const float4*>(rt);
  const int it = tneg[(size_t)b * K_NEG + tid];
  const int ih = hneg[(size_t)b * K_NEG + tid];
  const float4* xt4 = reinterpret_cast<const float4*>(x + (size_t)it * DIM);
  const float4* xh4 = reinterpret_cast<const float4*>(x + (size_t)ih * DIM);
  float st = 0.f, sh = 0.f;
#pragma unroll 8
  for (int i = 0; i < 32; ++i) {
    const float4 a = hr4[i], bt = xt4[i];
    st += a.x * bt.x + a.y * bt.y + a.z * bt.z + a.w * bt.w;
    const float4 c = rt4[i], dh = xh4[i];
    sh += c.x * dh.x + c.y * dh.y + c.z * dh.z + c.w * dh.w;
  }
  out[(size_t)b * 513 + 1 + tid]   = st;
  out[(size_t)b * 513 + 257 + tid] = sh;
}

// ---------------- host launch ----------------

extern "C" void kernel_launch(void* const* d_in, const int* in_sizes, int n_in,
                              void* d_out, int out_size, void* d_ws, size_t ws_size,
                              hipStream_t stream) {
  const float* entity  = (const float*)d_in[0];
  const float* self_W  = (const float*)d_in[1];
  const float* rel_W   = (const float*)d_in[2];
  const float* ln_g    = (const float*)d_in[3];
  const float* ln_b    = (const float*)d_in[4];
  const float* rel_dec = (const float*)d_in[5];
  const int*   src     = (const int*)d_in[6];
  const int*   dst     = (const int*)d_in[7];
  const int*   h       = (const int*)d_in[8];
  const int*   r       = (const int*)d_in[9];
  const int*   t       = (const int*)d_in[10];
  const int*   tneg    = (const int*)d_in[11];
  const int*   hneg    = (const int*)d_in[12];
  float* out = (float*)d_out;

  char* ws = (char*)d_ws;
  const size_t szX  = (size_t)N_ENT * DIM * sizeof(float);          // 51.2 MB
  const size_t szXB = (size_t)N_ENT * DIM * sizeof(unsigned short); // 25.6 MB
  float*          buf0 = (float*)(ws);
  float*          buf1 = (float*)(ws + szX);
  unsigned short* xb   = (unsigned short*)(ws + 2 * szX);
  unsigned short* wtb  = (unsigned short*)(ws + 2 * szX + szXB);    // 2*9*128*128 bf16

  const int nelX = N_ENT * DIM;                 // 12,800,000
  const int nelW = 2 * 9 * DIM * DIM;           // 294,912

  cvt_w_bf16<<<nelW / 256, 256, 0, stream>>>(self_W, rel_W, wtb);
  cvt_x_bf16<<<(nelX + 255) / 256, 256, 0, stream>>>(entity, xb, nelX);

  const dim3 msgGrid((ETILES + TILES_PER_BLK - 1) / TILES_PER_BLK, N_REL);  // (1563, 8)
  const int  gemmBlocks = (N_ENT + 63) / 64;    // 1563
  const int  lnBlocks   = (N_ENT + 7) / 8;      // 12500

  // layer 0: x_cur = entity, out/next = buf0
  self_gemm<<<gemmBlocks, 256, 0, stream>>>(xb, wtb + 0 * 9 * 16384, buf0, N_ENT);
  msg_gemm<<<msgGrid, 256, 0, stream>>>(xb, wtb + (0 * 9 + 1) * 16384, src, dst, buf0);
  ln_residual<<<lnBlocks, 256, 0, stream>>>(entity, buf0, xb, ln_g + 0 * DIM, ln_b + 0 * DIM,
                                            N_ENT, /*relu=*/1);
  // layer 1: x_cur = buf0, out/next = buf1
  self_gemm<<<gemmBlocks, 256, 0, stream>>>(xb, wtb + 1 * 9 * 16384, buf1, N_ENT);
  msg_gemm<<<msgGrid, 256, 0, stream>>>(xb, wtb + (1 * 9 + 1) * 16384, src, dst, buf1);
  ln_residual<<<lnBlocks, 256, 0, stream>>>(buf0, buf1, xb, ln_g + 1 * DIM, ln_b + 1 * DIM,
                                            N_ENT, /*relu=*/0);

  decode_scores<<<BATCH, K_NEG, 0, stream>>>(buf1, rel_dec, h, r, t, tneg, hneg, out);
}
